// LeanFFN_29308856828605
// MI455X (gfx1250) — compile-verified, matrix operand order checked by reading, not run on hardware
//
#include <hip/hip_runtime.h>
#include <hip/hip_bf16.h>

// ---------------------------------------------------------------------------
// LeanFFN for MI455X (gfx1250): LN -> GEMM1+GeGLU -> GEMM2+LN+residual
// bf16 WMMA (v_wmma_f32_16x16x32_bf16) with f32 accumulation, wave32.
// Register-blocked 2x in M so both GEMMs run at 2 b128 loads per WMMA.
// ---------------------------------------------------------------------------

typedef __attribute__((ext_vector_type(16))) __bf16 v16bf;
typedef __attribute__((ext_vector_type(8)))  __bf16 v8bf;
typedef __attribute__((ext_vector_type(8)))  float  v8f;

#define HDIM 1024
#define IDIM 4096
#define NROWS 8192
#define LN_EPS 1e-5f

// Load one WMMA 16x32 bf16 operand (A: MxK row-major, or B-as-N-rows: each
// lane n reads W row n contiguously in K). Per the CDNA5 ISA 16-bit operand
// layout: lanes 0-15 hold K {0..7, 16..23}, lanes 16-31 hold K {8..15, 24..31}.
// Two global_load_b128 per lane.
__device__ __forceinline__ v16bf load_op16x32(const __bf16* __restrict__ base,
                                              int ld, int k0, int lane) {
    int r  = lane & 15;
    int kh = (lane >> 4) << 3;           // 0 or 8
    const __bf16* p = base + (size_t)r * ld + k0 + kh;
    v8bf lo = *(const v8bf*)(p);         // K = k0+kh .. +7
    v8bf hi = *(const v8bf*)(p + 16);    // K = k0+16+kh .. +7
    return __builtin_shufflevector(lo, hi, 0, 1, 2, 3, 4, 5, 6, 7,
                                           8, 9, 10, 11, 12, 13, 14, 15);
}

// ---------------------------------------------------------------------------
// Kernel 0: fp32 -> bf16 weight conversion (4 elements / thread)
// ---------------------------------------------------------------------------
__global__ void __launch_bounds__(256)
cvt_bf16_kernel(const float* __restrict__ src, __hip_bfloat16* __restrict__ dst, int n4) {
    int i = blockIdx.x * blockDim.x + threadIdx.x;
    if (i >= n4) return;
    float4 v = ((const float4*)src)[i];
    int o = i * 4;
    dst[o + 0] = __float2bfloat16(v.x);
    dst[o + 1] = __float2bfloat16(v.y);
    dst[o + 2] = __float2bfloat16(v.z);
    dst[o + 3] = __float2bfloat16(v.w);
}

// ---------------------------------------------------------------------------
// Kernel 1: per-row LayerNorm of x, cast to bf16. One block (256 thr) per row.
// ---------------------------------------------------------------------------
__global__ void __launch_bounds__(256)
ln_cast_kernel(const float* __restrict__ x, const float* __restrict__ w,
               const float* __restrict__ b, __hip_bfloat16* __restrict__ out) {
    int row = blockIdx.x;
    int t   = threadIdx.x;                         // 0..255, 4 elems each
    const float* xr = x + (size_t)row * HDIM;
    float4 v = ((const float4*)xr)[t];
    float s  = v.x + v.y + v.z + v.w;
    float s2 = v.x * v.x + v.y * v.y + v.z * v.z + v.w * v.w;
    #pragma unroll
    for (int m = 16; m >= 1; m >>= 1) {
        s  += __shfl_xor(s,  m);
        s2 += __shfl_xor(s2, m);
    }
    __shared__ float ls[8], ls2[8];
    int wv = t >> 5;
    if ((t & 31) == 0) { ls[wv] = s; ls2[wv] = s2; }
    __syncthreads();
    if (t == 0) {
        float a = 0.f, c = 0.f;
        #pragma unroll
        for (int i = 0; i < 8; ++i) { a += ls[i]; c += ls2[i]; }
        ls[0] = a; ls2[0] = c;
    }
    __syncthreads();
    float mu  = ls[0] * (1.f / HDIM);
    float var = ls2[0] * (1.f / HDIM) - mu * mu;
    float inv = rsqrtf(var + LN_EPS);
    int c0 = t * 4;
    __hip_bfloat16* o = out + (size_t)row * HDIM + c0;
    o[0] = __float2bfloat16((v.x - mu) * inv * w[c0 + 0] + b[c0 + 0]);
    o[1] = __float2bfloat16((v.y - mu) * inv * w[c0 + 1] + b[c0 + 1]);
    o[2] = __float2bfloat16((v.z - mu) * inv * w[c0 + 2] + b[c0 + 2]);
    o[3] = __float2bfloat16((v.w - mu) * inv * w[c0 + 3] + b[c0 + 3]);
}

// ---------------------------------------------------------------------------
// Kernel 2: GEMM1 (x_ln @ W1^T) fused with bias + GeGLU. Each wave computes
// TWO 16-row tiles of one (gate,lin) column pair: 4 accumulators, so each
// K-step is 8 global_load_b128 feeding 4 WMMAs (2 loads/WMMA). GeGLU fuses
// in registers; `pre` (256 MB) is never materialized.
// grid = (IDIM/128, NROWS/32), block = 256 (8 waves across 128 hid cols).
// ---------------------------------------------------------------------------
__global__ void __launch_bounds__(256)
gemm1_geglu_kernel(const __bf16* __restrict__ xln, const __bf16* __restrict__ w1,
                   const float* __restrict__ i2h_bias, __hip_bfloat16* __restrict__ hid) {
    int lane = threadIdx.x & 31;
    int wave = threadIdx.x >> 5;
    int m0 = blockIdx.y * 32;
    int jc = blockIdx.x * 128 + wave * 16;         // hid column tile base

    const __bf16* A0 = xln + (size_t)m0 * HDIM;
    const __bf16* A1 = xln + (size_t)(m0 + 16) * HDIM;
    const __bf16* Bg = w1 + (size_t)jc * HDIM;               // gate channels
    const __bf16* Bl = w1 + (size_t)(jc + IDIM) * HDIM;      // lin channels

    v8f accg[2] = {}; v8f accl[2] = {};
    for (int k0 = 0; k0 < HDIM; k0 += 32) {
        v16bf a0 = load_op16x32(A0, HDIM, k0, lane);
        v16bf a1 = load_op16x32(A1, HDIM, k0, lane);
        v16bf bg = load_op16x32(Bg, HDIM, k0, lane);
        v16bf bl = load_op16x32(Bl, HDIM, k0, lane);
        accg[0] = __builtin_amdgcn_wmma_f32_16x16x32_bf16(false, a0, false, bg,
                                                          (short)0, accg[0], false, false);
        accl[0] = __builtin_amdgcn_wmma_f32_16x16x32_bf16(false, a0, false, bl,
                                                          (short)0, accl[0], false, false);
        accg[1] = __builtin_amdgcn_wmma_f32_16x16x32_bf16(false, a1, false, bg,
                                                          (short)0, accg[1], false, false);
        accl[1] = __builtin_amdgcn_wmma_f32_16x16x32_bf16(false, a1, false, bl,
                                                          (short)0, accl[1], false, false);
    }

    int col = jc + (lane & 15);
    float bgs = i2h_bias[col];
    float bls = i2h_bias[col + IDIM];
    int rb = (lane >> 4) << 3;                     // C/D layout: lanes>=16 hold M+8
    #pragma unroll
    for (int mt = 0; mt < 2; ++mt) {
        int rbase = m0 + mt * 16 + rb;
        #pragma unroll
        for (int r = 0; r < 8; ++r) {
            float g = accg[mt][r] + bgs;
            float l = accl[mt][r] + bls;
            float u = 0.7978845608028654f * (g + 0.044715f * g * g * g);
            float h = 0.5f * g * (1.0f + tanhf(u)) * l;     // tanh-approx GELU gate
            hid[(size_t)(rbase + r) * IDIM + col] = __float2bfloat16(h);
        }
    }
}

// ---------------------------------------------------------------------------
// Kernel 3: GEMM2 (hid @ W2^T) + h2o_bias + post-LN + residual, fully fused.
// One block owns 32 complete rows x 1024 cols: 32 waves, each wave computes a
// 2x2 grid of 16x16 tiles (32 rows x 32 cols), so each K-step is 8 loads for
// 4 WMMAs. Row stats for the post-LN via shfl halves + ds_add_f32 into LDS.
// grid = NROWS/32, block = 1024.
// ---------------------------------------------------------------------------
__global__ void __launch_bounds__(1024)
gemm2_postln_kernel(const __bf16* __restrict__ hid, const __bf16* __restrict__ w2,
                    const float* __restrict__ h2o_bias, const float* __restrict__ pw,
                    const float* __restrict__ pb, const float* __restrict__ x,
                    float* __restrict__ out) {
    __shared__ float s_sum[32], s_sq[32];
    int lane = threadIdx.x & 31;
    int wave = threadIdx.x >> 5;
    int m0 = blockIdx.x * 32;
    int n0 = wave * 32;

    const __bf16* A0 = hid + (size_t)m0 * IDIM;
    const __bf16* A1 = hid + (size_t)(m0 + 16) * IDIM;
    const __bf16* B0 = w2 + (size_t)n0 * IDIM;
    const __bf16* B1 = w2 + (size_t)(n0 + 16) * IDIM;

    v8f acc[2][2] = {};                            // [m tile][n tile]
    for (int k0 = 0; k0 < IDIM; k0 += 32) {
        v16bf a0 = load_op16x32(A0, IDIM, k0, lane);
        v16bf a1 = load_op16x32(A1, IDIM, k0, lane);
        v16bf b0 = load_op16x32(B0, IDIM, k0, lane);
        v16bf b1 = load_op16x32(B1, IDIM, k0, lane);
        acc[0][0] = __builtin_amdgcn_wmma_f32_16x16x32_bf16(false, a0, false, b0,
                                                            (short)0, acc[0][0], false, false);
        acc[0][1] = __builtin_amdgcn_wmma_f32_16x16x32_bf16(false, a0, false, b1,
                                                            (short)0, acc[0][1], false, false);
        acc[1][0] = __builtin_amdgcn_wmma_f32_16x16x32_bf16(false, a1, false, b0,
                                                            (short)0, acc[1][0], false, false);
        acc[1][1] = __builtin_amdgcn_wmma_f32_16x16x32_bf16(false, a1, false, b1,
                                                            (short)0, acc[1][1], false, false);
    }

    if (threadIdx.x < 32) { s_sum[threadIdx.x] = 0.f; s_sq[threadIdx.x] = 0.f; }
    __syncthreads();

    int c0 = n0 + (lane & 15);
    int c1 = c0 + 16;
    float bb0 = h2o_bias[c0];
    float bb1 = h2o_bias[c1];
    int rb = (lane >> 4) << 3;                     // 0 or 8

    #pragma unroll
    for (int mt = 0; mt < 2; ++mt) {
        #pragma unroll
        for (int r = 0; r < 8; ++r) {
            acc[mt][0][r] += bb0;
            acc[mt][1][r] += bb1;
            float p = acc[mt][0][r] + acc[mt][1][r];
            float q = acc[mt][0][r] * acc[mt][0][r] + acc[mt][1][r] * acc[mt][1][r];
            #pragma unroll
            for (int m = 8; m >= 1; m >>= 1) {      // reduce within 16-lane half
                p += __shfl_xor(p, m);
                q += __shfl_xor(q, m);
            }
            if ((lane & 15) == 0) {                 // lanes 0 & 16: rows r, r+8
                atomicAdd(&s_sum[mt * 16 + rb + r], p);   // -> ds_add_f32
                atomicAdd(&s_sq[mt * 16 + rb + r],  q);
            }
        }
    }
    __syncthreads();

    #pragma unroll
    for (int mt = 0; mt < 2; ++mt) {
        #pragma unroll
        for (int r = 0; r < 8; ++r) {
            int row = mt * 16 + rb + r;
            float mu  = s_sum[row] * (1.f / HDIM);
            float var = s_sq[row] * (1.f / HDIM) - mu * mu;
            float inv = rsqrtf(var + LN_EPS);
            size_t grow = (size_t)(m0 + row);
            out[grow * HDIM + c0] = (acc[mt][0][r] - mu) * inv * pw[c0] + pb[c0]
                                    + x[grow * HDIM + c0];
            out[grow * HDIM + c1] = (acc[mt][1][r] - mu) * inv * pw[c1] + pb[c1]
                                    + x[grow * HDIM + c1];
        }
    }
}

// ---------------------------------------------------------------------------
// Host launch
// ---------------------------------------------------------------------------
extern "C" void kernel_launch(void* const* d_in, const int* in_sizes, int n_in,
                              void* d_out, int out_size, void* d_ws, size_t ws_size,
                              hipStream_t stream) {
    const float* x    = (const float*)d_in[0];
    const float* ln_w = (const float*)d_in[1];
    const float* ln_b = (const float*)d_in[2];
    const float* w1   = (const float*)d_in[3];   // (2I, H) = (8192, 1024)
    const float* b1   = (const float*)d_in[4];   // (8192,)
    const float* w2   = (const float*)d_in[5];   // (H, I) = (1024, 4096)
    const float* b2   = (const float*)d_in[6];   // (1024,)
    const float* pw   = (const float*)d_in[7];
    const float* pb   = (const float*)d_in[8];
    float* out = (float*)d_out;

    // Workspace layout (bytes):
    //   [0,16MB)   x_ln bf16 (8192 x 1024)
    //   [16,32MB)  W1 bf16   (8192 x 1024)
    //   [32,40MB)  W2 bf16   (1024 x 4096)
    //   [40,104MB) hid bf16  (8192 x 4096)
    char* ws = (char*)d_ws;
    __hip_bfloat16* xln = (__hip_bfloat16*)(ws);
    __hip_bfloat16* w1b = (__hip_bfloat16*)(ws + (size_t)16 * 1024 * 1024);
    __hip_bfloat16* w2b = (__hip_bfloat16*)(ws + (size_t)32 * 1024 * 1024);
    __hip_bfloat16* hid = (__hip_bfloat16*)(ws + (size_t)40 * 1024 * 1024);

    // 0) weight conversions
    {
        int n4 = (2 * IDIM * HDIM) / 4;            // 8192*1024 / 4
        cvt_bf16_kernel<<<(n4 + 255) / 256, 256, 0, stream>>>(w1, w1b, n4);
        int m4 = (HDIM * IDIM) / 4;                // 1024*4096 / 4
        cvt_bf16_kernel<<<(m4 + 255) / 256, 256, 0, stream>>>(w2, w2b, m4);
    }
    // 1) LN + cast
    ln_cast_kernel<<<NROWS, 256, 0, stream>>>(x, ln_w, ln_b, xln);
    // 2) GEMM1 + GeGLU
    gemm1_geglu_kernel<<<dim3(IDIM / 128, NROWS / 32), 256, 0, stream>>>(
        (const __bf16*)xln, (const __bf16*)w1b, b1, hid);
    // 3) GEMM2 + post-LN + residual
    gemm2_postln_kernel<<<NROWS / 32, 1024, 0, stream>>>(
        (const __bf16*)hid, (const __bf16*)w2b, b2, pw, pb, x, out);
}
